// Encoder_21646635172361
// MI455X (gfx1250) — matrix operand order repaired and checked by reading, not compiled, not run on hardware
//
#include <hip/hip_runtime.h>

// GCNConv (N=50000, E=400000, F=128, H=512) + PReLU for MI455X (gfx1250).
//
// Strategy (reasoned from MI455X specs, no runtime available):
//  * GEMM h = x @ W^T done in bf16 WMMA (v_wmma_f32_16x16x32_bf16, f32 accum).
//    6.55 GFLOP -> negligible next to the edge phase once on the matrix path.
//    One cheap conversion pass (38 MB ~ 1.6us @ 23.3 TB/s) enables it.
//  * h (102.4 MB) lives in d_ws and fits the 192 MB L2, so the random
//    per-edge gather (450k x 2KB) + f32 atomic scatter stays L2-resident.
//  * All dims divide evenly -> no tails, EXEC all-1s in the WMMA kernel.

#define N_NODES 50000
#define N_EDGES 400000
#define F_IN    128
#define H_OUT   512

typedef __attribute__((ext_vector_type(16))) __bf16 v16bf;
typedef __attribute__((ext_vector_type(8)))  float  v8f;

union ABFrag {
    v16bf v;
    uint4 u[2];
};

// ---------------------------------------------------------------------------
// Utility: zero an f32 buffer (float4 vectorized; n4 = count/4, exact here)
// ---------------------------------------------------------------------------
__global__ void zero_f32(float* __restrict__ p, int n4) {
    int i = blockIdx.x * blockDim.x + threadIdx.x;
    if (i < n4) {
        ((float4*)p)[i] = make_float4(0.f, 0.f, 0.f, 0.f);
    }
}

// ---------------------------------------------------------------------------
// f32 -> bf16 conversion (float4 in, 4x bf16 = 8 bytes out)
// ---------------------------------------------------------------------------
__global__ void f32_to_bf16(const float* __restrict__ in,
                            __bf16* __restrict__ out, int n4) {
    int i = blockIdx.x * blockDim.x + threadIdx.x;
    if (i >= n4) return;
    float4 v = ((const float4*)in)[i];
    union { __bf16 h[4]; uint2 u; } cv;
    cv.h[0] = (__bf16)v.x;
    cv.h[1] = (__bf16)v.y;
    cv.h[2] = (__bf16)v.z;
    cv.h[3] = (__bf16)v.w;
    ((uint2*)out)[i] = cv.u;
}

// ---------------------------------------------------------------------------
// Degree accumulation at edge targets (reference: segment_sum of ones at col)
// ---------------------------------------------------------------------------
__global__ void deg_accum(const long long* __restrict__ ei,
                          float* __restrict__ deg) {
    int e = blockIdx.x * blockDim.x + threadIdx.x;
    if (e >= N_EDGES) return;
    int c = (int)ei[N_EDGES + e];   // edge_index[1][e] (int64 row-major [2,E])
    atomicAdd(&deg[c], 1.0f);
}

// deg -> dinv = 1/sqrt(deg + 1)  (self-loop contributes +1, so deg_total >= 1)
__global__ void make_dinv(float* __restrict__ deg) {
    int i = blockIdx.x * blockDim.x + threadIdx.x;
    if (i >= N_NODES) return;
    deg[i] = rsqrtf(deg[i] + 1.0f);
}

// ---------------------------------------------------------------------------
// WMMA GEMM: h[N,H] = xb[N,F] * wb[H,F]^T  (bf16 inputs, f32 accumulate)
// Block = 256 threads = 8 waves. Block tile = 16 rows x 128 cols;
// each wave owns one 16x16 output tile, K=128 in 4 steps of 32.
//
// A-fragment (16x32 bf16, ISA 7.12.2 layout): lane L holds row M = L&15;
//   lanes 0-15:  VGPR0-3 = K 0..7,   VGPR4-7 = K 16..23
//   lanes 16-31: VGPR0-3 = K 8..15,  VGPR4-7 = K 24..31
// -> per lane: two contiguous 16B (8 x bf16) loads from the row.
// B-fragment (32x16 bf16): lane holds col N = L&15; K order same as A:
// lanes 0-15 K 0..7,16..23; lanes 16-31 K 8..15,24..31 -> two 16B loads per lane
// from wb row (wb is [H,F] row-major, i.e. K-contiguous per output channel).
// ---------------------------------------------------------------------------
__global__ __launch_bounds__(256) void gemm_wmma_bf16(
    const __bf16* __restrict__ xb, const __bf16* __restrict__ wb,
    float* __restrict__ h) {
    const int lane = threadIdx.x & 31;
    const int wave = threadIdx.x >> 5;
    const int m0 = blockIdx.x * 16;
    const int n0 = blockIdx.y * 128 + wave * 16;

    const int am    = lane & 15;
    const int akoff = (lane & 16) ? 8 : 0;
    const int bn    = lane & 15;
    const int bkoff = (lane & 16) ? 8 : 0;

    const __bf16* arow = xb + (size_t)(m0 + am) * F_IN;
    const __bf16* brow = wb + (size_t)(n0 + bn) * F_IN;

    v8f c = {0.f, 0.f, 0.f, 0.f, 0.f, 0.f, 0.f, 0.f};
#pragma unroll
    for (int kc = 0; kc < F_IN; kc += 32) {
        ABFrag a, b;
        a.u[0] = *(const uint4*)(arow + kc + akoff);        // K kc+akoff .. +7
        a.u[1] = *(const uint4*)(arow + kc + 16 + akoff);   // K kc+16+akoff .. +7
        b.u[0] = *(const uint4*)(brow + kc + bkoff);        // K kc+bkoff .. +7
        b.u[1] = *(const uint4*)(brow + kc + 16 + bkoff);   // K kc+16+bkoff .. +7
        c = __builtin_amdgcn_wmma_f32_16x16x32_bf16(
                /*neg_a=*/false, a.v, /*neg_b=*/false, b.v,
                /*c_mod=*/(short)0, c, /*reuse_a=*/false, /*reuse_b=*/false);
    }

    // C/D layout: VGPR i -> M = i (lanes 0-15) or i+8 (lanes 16-31), N = lane&15
    const int row0 = (lane & 16) ? 8 : 0;
    const int col  = lane & 15;
    float* out = h + (size_t)(m0 + row0) * H_OUT + (n0 + col);
#pragma unroll
    for (int i = 0; i < 8; ++i)
        out[(size_t)i * H_OUT] = c[i];
}

// ---------------------------------------------------------------------------
// Edge scatter: out[col] += h[row] * dinv[row]*dinv[col]
// 2 edges per 256-thread block; 128 threads/edge, float4 per thread.
// h-row gathers and the f32 atomics stay L2-resident (h = 102 MB < 192 MB L2).
// ---------------------------------------------------------------------------
__global__ __launch_bounds__(256) void scatter_edges(
    const long long* __restrict__ ei, const float* __restrict__ h,
    const float* __restrict__ dinv, float* __restrict__ acc) {
    int e = blockIdx.x * 2 + (threadIdx.x >> 7);
    if (e >= N_EDGES) return;
    int t = threadIdx.x & 127;                // feature chunk (4 floats)
    int r = (int)ei[e];                       // source
    int c = (int)ei[N_EDGES + e];             // target
    float nrm = dinv[r] * dinv[c];
    float4 v = ((const float4*)(h + (size_t)r * H_OUT))[t];
    float* ap = acc + (size_t)c * H_OUT + t * 4;
    atomicAdd(ap + 0, v.x * nrm);
    atomicAdd(ap + 1, v.y * nrm);
    atomicAdd(ap + 2, v.z * nrm);
    atomicAdd(ap + 3, v.w * nrm);
}

// ---------------------------------------------------------------------------
// Finalize: out = PReLU(acc + h*dinv^2 (self-loop) + b)
// ---------------------------------------------------------------------------
__global__ void finalize(float* __restrict__ out, const float* __restrict__ h,
                         const float* __restrict__ dinv,
                         const float* __restrict__ b,
                         const float* __restrict__ alpha) {
    int i = blockIdx.x * blockDim.x + threadIdx.x;   // over N*H/4
    if (i >= N_NODES * (H_OUT / 4)) return;
    int row = i / (H_OUT / 4);
    int cc  = i % (H_OUT / 4);
    float di = dinv[row];
    float self = di * di;
    float4 a  = ((const float4*)out)[i];
    float4 hv = ((const float4*)h)[i];
    float4 bv = ((const float4*)b)[cc];
    float4 av = ((const float4*)alpha)[cc];
    float4 r;
    r.x = a.x + hv.x * self + bv.x;
    r.y = a.y + hv.y * self + bv.y;
    r.z = a.z + hv.z * self + bv.z;
    r.w = a.w + hv.w * self + bv.w;
    r.x = r.x > 0.f ? r.x : av.x * r.x;
    r.y = r.y > 0.f ? r.y : av.y * r.y;
    r.z = r.z > 0.f ? r.z : av.z * r.z;
    r.w = r.w > 0.f ? r.w : av.w * r.w;
    ((float4*)out)[i] = r;
}

// ---------------------------------------------------------------------------
extern "C" void kernel_launch(void* const* d_in, const int* in_sizes, int n_in,
                              void* d_out, int out_size, void* d_ws,
                              size_t ws_size, hipStream_t stream) {
    const float*     x     = (const float*)d_in[0];
    const long long* ei    = (const long long*)d_in[1];  // int64 [2, E]
    const float*     W     = (const float*)d_in[2];      // [H, F]
    const float*     b     = (const float*)d_in[3];      // [H]
    const float*     alpha = (const float*)d_in[4];      // [H]
    float*           out   = (float*)d_out;              // [N, H]

    // Workspace layout (all 16B-aligned sizes):
    //   h:    N*H f32   = 102,400,000 B
    //   xb:   N*F bf16  =  12,800,000 B
    //   wb:   H*F bf16  =     131,072 B
    //   dinv: N   f32   =     200,000 B
    char*   ws   = (char*)d_ws;
    float*  h    = (float*)ws;
    size_t  off  = (size_t)N_NODES * H_OUT * sizeof(float);
    __bf16* xb   = (__bf16*)(ws + off); off += (size_t)N_NODES * F_IN * 2;
    __bf16* wb   = (__bf16*)(ws + off); off += (size_t)H_OUT * F_IN * 2;
    float*  dinv = (float*)(ws + off);

    const int NH4 = N_NODES * (H_OUT / 4);   // 6,400,000
    const int NF4 = N_NODES * (F_IN / 4);    // 1,600,000
    const int WF4 = H_OUT * (F_IN / 4);      // 65,536... wait: 512*32 = 16,384

    zero_f32<<<(NH4 + 255) / 256, 256, 0, stream>>>(out, NH4);
    zero_f32<<<(N_NODES / 4 + 255) / 256, 256, 0, stream>>>(dinv, N_NODES / 4);

    f32_to_bf16<<<(NF4 + 255) / 256, 256, 0, stream>>>(x, xb, NF4);
    f32_to_bf16<<<(WF4 + 255) / 256, 256, 0, stream>>>(W, wb, WF4);

    deg_accum<<<(N_EDGES + 255) / 256, 256, 0, stream>>>(ei, dinv);
    make_dinv<<<(N_NODES + 255) / 256, 256, 0, stream>>>(dinv);

    gemm_wmma_bf16<<<dim3(N_NODES / 16, H_OUT / 128), 256, 0, stream>>>(xb, wb, h);

    scatter_edges<<<(N_EDGES + 1) / 2, 256, 0, stream>>>(ei, h, dinv, out);

    finalize<<<(NH4 + 255) / 256, 256, 0, stream>>>(out, h, dinv, b, alpha);
}
